// MultiHeadAttention_94489281267
// MI455X (gfx1250) — matrix operand order repaired and checked by reading, not compiled, not run on hardware
//
#include <hip/hip_runtime.h>

// ---------------------------------------------------------------------------
// MHA forward for MI455X (gfx1250): bf16 WMMA everywhere, f32 accumulate.
// B=2, S=2048, E=1024, H=16, D=64
// ---------------------------------------------------------------------------

#define Bn 2
#define Sn 2048
#define En 1024
#define Hn 16
#define Dn 64

typedef __attribute__((ext_vector_type(16))) __bf16 v16bf;
typedef __attribute__((ext_vector_type(8)))  float  v8f;
typedef __attribute__((ext_vector_type(4)))  unsigned int v4u;

union FragU { v4u u[2]; v16bf v; };

__device__ __forceinline__ v8f vzero8() {
  v8f z = {0.f, 0.f, 0.f, 0.f, 0.f, 0.f, 0.f, 0.f};
  return z;
}

__device__ __forceinline__ unsigned short f2bf(float f) {
  unsigned int u = __float_as_uint(f);
  u += 0x7FFFu + ((u >> 16) & 1u);          // round-to-nearest-even
  return (unsigned short)(u >> 16);
}

// A-fragment (M=16 x K=32, 16-bit). ISA layout: lanes 0-15 hold K {0..7,16..23},
// lanes 16-31 hold K {8..15,24..31}. rowp points at row start (+k0).
__device__ __forceinline__ v16bf load_fragA(const unsigned short* rowp, int lane) {
  const int kb8 = (lane >> 4) << 3;
  FragU f;
  f.u[0] = *reinterpret_cast<const v4u*>(rowp + kb8);
  f.u[1] = *reinterpret_cast<const v4u*>(rowp + kb8 + 16);
  return f.v;
}

// B-fragment (K=32 x N=16, 16-bit) loaded from transposed source Bt[n][k]:
// lanes 0-15 hold K {0..7,16..23}, lanes 16-31 hold K {8..15,24..31} (same as A).
// rowp points at Bt row for n = lane&15 (+k0).
__device__ __forceinline__ v16bf load_fragB(const unsigned short* rowp, int lane) {
  const int kb16 = (lane >> 4) << 3;
  FragU f;
  f.u[0] = *reinterpret_cast<const v4u*>(rowp + kb16);
  f.u[1] = *reinterpret_cast<const v4u*>(rowp + kb16 + 16);
  return f.v;
}

__device__ __forceinline__ v8f wmma_bf16(v16bf a, v16bf b, v8f c) {
  return __builtin_amdgcn_wmma_f32_16x16x32_bf16(
      /*neg_a=*/false, a, /*neg_b=*/false, b,
      /*c_mod=*/(short)0, c, /*reuse_a=*/false, /*reuse_b=*/false);
}

// ---------------------------------------------------------------------------
// Conversion kernels
// ---------------------------------------------------------------------------
__global__ void k_f32_to_bf16(const float* __restrict__ in,
                              unsigned short* __restrict__ out, int n) {
  int i = blockIdx.x * blockDim.x + threadIdx.x;
  if (i < n) out[i] = f2bf(in[i]);
}

// in: [K][N] fp32 row-major  ->  out: [N][K] bf16 (transposed)
__global__ void k_transpose_to_bf16(const float* __restrict__ in,
                                    unsigned short* __restrict__ out,
                                    int K, int N) {
  int i = blockIdx.x * blockDim.x + threadIdx.x;
  if (i < K * N) {
    int n = i / K, k = i - n * K;
    out[i] = f2bf(in[k * N + n]);
  }
}

// ---------------------------------------------------------------------------
// QKV GEMM: [4096,1024] x [1024,3072] -> scatter into Q, K (row-major [BH,S,D])
// and V transposed ([BH,D,S]). Q pre-scaled by D^-0.5.
// Wave tile: 64(M) x 64(N): 16 b128 loads : 16 WMMAs per K-step.
// ---------------------------------------------------------------------------
__global__ void __launch_bounds__(256)
k_gemm_qkv(const unsigned short* __restrict__ xb,
           const unsigned short* __restrict__ wt,   // [3E][E] transposed bf16
           unsigned short* __restrict__ Qb,
           unsigned short* __restrict__ Kb,
           unsigned short* __restrict__ Vt) {
  const int lane = threadIdx.x & 31;
  const int wave = blockIdx.x * (blockDim.x >> 5) + (threadIdx.x >> 5);
  const int tilesN = (3 * En) / 64;                 // 48
  const int mb = (wave / tilesN) * 64;
  const int nb = (wave % tilesN) * 64;
  const int hb8 = (lane >> 4) << 3;

  v8f acc[4][4];
  for (int im = 0; im < 4; ++im)
    for (int in = 0; in < 4; ++in) acc[im][in] = vzero8();

  for (int k0 = 0; k0 < En; k0 += 32) {
    v16bf a[4], b[4];
    for (int im = 0; im < 4; ++im)
      a[im] = load_fragA(xb + (mb + im * 16 + (lane & 15)) * En + k0, lane);
    for (int in = 0; in < 4; ++in)
      b[in] = load_fragB(wt + (nb + in * 16 + (lane & 15)) * En + k0, lane);
    for (int im = 0; im < 4; ++im)
      for (int in = 0; in < 4; ++in)
        acc[im][in] = wmma_bf16(a[im], b[in], acc[im][in]);
  }

  // Epilogue. mb is a multiple of 64 and S=2048 is a multiple of 64, so the
  // batch index and s-offset are wave-uniform.
  const int bidx = mb >> 11;
  const int s0   = mb & (Sn - 1);
  for (int in = 0; in < 4; ++in) {
    const int ntile = nb + in * 16;
    const int c = ntile >> 10;                      // 0=q 1=k 2=v (wave-uniform)
    const int h = (ntile >> 6) & (Hn - 1);
    const int d = (ntile & (Dn - 1)) + (lane & 15);
    const int head = bidx * Hn + h;
    if (c == 2) {
      unsigned short* dst = Vt + (size_t)(head * Dn + d) * Sn + s0;
      for (int im = 0; im < 4; ++im)
        for (int r = 0; r < 8; ++r)
          dst[im * 16 + r + hb8] = f2bf(acc[im][in][r]);
    } else {
      unsigned short* dst =
          (c == 0 ? Qb : Kb) + (size_t)head * Sn * Dn + (size_t)s0 * Dn + d;
      const float sc = (c == 0) ? 0.125f : 1.0f;    // D^-0.5 folded into Q
      for (int im = 0; im < 4; ++im)
        for (int r = 0; r < 8; ++r)
          dst[(im * 16 + r + hb8) * Dn] = f2bf(acc[im][in][r] * sc);
    }
  }
}

// ---------------------------------------------------------------------------
// Flash attention: one wave per block, 32 query rows (two 16-row tiles) per
// wave so every K/V fragment is reused twice. Causal loop over 32-wide key
// blocks; online softmax; P transposed via LDS.
// ---------------------------------------------------------------------------
__global__ void __launch_bounds__(32)
k_attention(const unsigned short* __restrict__ Qb,
            const unsigned short* __restrict__ Kb,
            const unsigned short* __restrict__ Vt,
            unsigned short* __restrict__ Ob) {
  __shared__ __align__(16) unsigned short ldsP[2 * 16 * 32];

  const int lane = threadIdx.x & 31;
  const int hb8  = (lane >> 4) << 3;                // 0 or 8
  const int qblk = blockIdx.x & (Sn / 32 - 1);      // 64 q-blocks per head
  const int bh   = blockIdx.x >> 6;
  const int qb   = qblk * 32;

  const unsigned short* Qhead = Qb + (size_t)bh * Sn * Dn;
  const unsigned short* Khead = Kb + (size_t)bh * Sn * Dn;
  const unsigned short* Vhead = Vt + (size_t)bh * Dn * Sn;

  v16bf aq[2][2];
  for (int qt = 0; qt < 2; ++qt) {
    const unsigned short* qrow = Qhead + (qb + qt * 16 + (lane & 15)) * Dn;
    aq[qt][0] = load_fragA(qrow, lane);
    aq[qt][1] = load_fragA(qrow + 32, lane);
  }

  float m_i[2][8], l_i[2][8];
  v8f o[2][4];
  for (int qt = 0; qt < 2; ++qt) {
    for (int r = 0; r < 8; ++r) { m_i[qt][r] = -1e30f; l_i[qt][r] = 0.f; }
    for (int t = 0; t < 4; ++t) o[qt][t] = vzero8();
  }

  for (int skb = 0; skb < qb + 32; skb += 32) {
    // ---- K fragments for this 32-key block (shared by both q tiles)
    v16bf bk[2][2];
    for (int st = 0; st < 2; ++st) {
      const unsigned short* krow = Khead + (skb + st * 16 + (lane & 15)) * Dn;
      bk[st][0] = load_fragB(krow, lane);
      bk[st][1] = load_fragB(krow + 32, lane);
    }
    for (int qt = 0; qt < 2; ++qt) {
      // ---- scores: two 16x16 tiles, K-dim = D = 64
      v8f s[2];
      for (int st = 0; st < 2; ++st) {
        v8f z = vzero8();
        z = wmma_bf16(aq[qt][0], bk[st][0], z);
        z = wmma_bf16(aq[qt][1], bk[st][1], z);
        s[st] = z;
      }
      // ---- causal mask + online softmax (rows live in 16-lane halves)
      float p0[8], p1[8], alpha[8];
      for (int r = 0; r < 8; ++r) {
        const int sq  = qb + qt * 16 + r + hb8;
        const int sk0 = skb + (lane & 15);
        const float v0 = (sk0      <= sq) ? s[0][r] : -1e30f;
        const float v1 = (sk0 + 16 <= sq) ? s[1][r] : -1e30f;
        float rm = fmaxf(v0, v1);
        for (int msk = 1; msk < 16; msk <<= 1)
          rm = fmaxf(rm, __shfl_xor(rm, msk, 32));
        const float mn = fmaxf(m_i[qt][r], rm);
        alpha[r] = __expf(m_i[qt][r] - mn);
        p0[r] = __expf(v0 - mn);
        p1[r] = __expf(v1 - mn);
        float rs = p0[r] + p1[r];
        for (int msk = 1; msk < 16; msk <<= 1)
          rs += __shfl_xor(rs, msk, 32);
        l_i[qt][r] = l_i[qt][r] * alpha[r] + rs;
        m_i[qt][r] = mn;
      }
      for (int t = 0; t < 4; ++t)
        for (int r = 0; r < 8; ++r)
          o[qt][t][r] *= alpha[r];
      // ---- transpose P (C-layout -> A-fragment layout) through LDS
      unsigned short* pbuf = ldsP + qt * (16 * 32);
      for (int r = 0; r < 8; ++r) {
        const int row = r + hb8;
        pbuf[row * 32 + (lane & 15)]      = f2bf(p0[r]);
        pbuf[row * 32 + 16 + (lane & 15)] = f2bf(p1[r]);
      }
    }
    __syncthreads();   // single-wave WG: cheap; DS ops are wave-in-order
    // ---- V fragments for this key block (shared by both q tiles)
    v16bf bv[4];
    for (int t = 0; t < 4; ++t)
      bv[t] = load_fragB(Vhead + (t * 16 + (lane & 15)) * Sn + skb, lane);
    for (int qt = 0; qt < 2; ++qt) {
      const v16bf ap = load_fragA(
          (const unsigned short*)ldsP + qt * (16 * 32) + (lane & 15) * 32, lane);
      for (int t = 0; t < 4; ++t)
        o[qt][t] = wmma_bf16(ap, bv[t], o[qt][t]);
    }
    __syncthreads();
  }

  // ---- normalize, store to Ob[b*S+sq][h*D + d] (bf16, input of proj GEMM)
  const int bidx = bh >> 4;
  const int h    = bh & (Hn - 1);
  for (int qt = 0; qt < 2; ++qt) {
    for (int r = 0; r < 8; ++r) {
      const float inv = 1.f / l_i[qt][r];
      const int sq = qb + qt * 16 + r + hb8;
      const size_t rowo = ((size_t)bidx * Sn + sq) * En + h * Dn + (lane & 15);
      for (int t = 0; t < 4; ++t)
        Ob[rowo + t * 16] = f2bf(o[qt][t][r] * inv);
    }
  }
}

// ---------------------------------------------------------------------------
// Projection GEMM: [4096,1024] x [1024,1024] -> fp32 out. 64x64 per wave.
// ---------------------------------------------------------------------------
__global__ void __launch_bounds__(256)
k_gemm_proj(const unsigned short* __restrict__ Ab,
            const unsigned short* __restrict__ wt,  // [E][E] transposed bf16
            float* __restrict__ out) {
  const int lane = threadIdx.x & 31;
  const int wave = blockIdx.x * (blockDim.x >> 5) + (threadIdx.x >> 5);
  const int tilesN = En / 64;                       // 16
  const int mb = (wave / tilesN) * 64;
  const int nb = (wave % tilesN) * 64;
  const int hb8 = (lane >> 4) << 3;

  v8f acc[4][4];
  for (int im = 0; im < 4; ++im)
    for (int in = 0; in < 4; ++in) acc[im][in] = vzero8();

  for (int k0 = 0; k0 < En; k0 += 32) {
    v16bf a[4], b[4];
    for (int im = 0; im < 4; ++im)
      a[im] = load_fragA(Ab + (mb + im * 16 + (lane & 15)) * En + k0, lane);
    for (int in = 0; in < 4; ++in)
      b[in] = load_fragB(wt + (nb + in * 16 + (lane & 15)) * En + k0, lane);
    for (int im = 0; im < 4; ++im)
      for (int in = 0; in < 4; ++in)
        acc[im][in] = wmma_bf16(a[im], b[in], acc[im][in]);
  }

  for (int im = 0; im < 4; ++im) {
    for (int in = 0; in < 4; ++in) {
      float* dst = out + (size_t)(mb + im * 16 + hb8) * En + nb + in * 16 + (lane & 15);
      for (int r = 0; r < 8; ++r)
        dst[(size_t)r * En] = acc[im][in][r];
    }
  }
}

// ---------------------------------------------------------------------------
// Host launch
// ---------------------------------------------------------------------------
extern "C" void kernel_launch(void* const* d_in, const int* in_sizes, int n_in,
                              void* d_out, int out_size, void* d_ws, size_t ws_size,
                              hipStream_t stream) {
  const float* x     = (const float*)d_in[0];
  // d_in[1] = attention_mask (causal tril) — implemented analytically
  const float* Wqkv  = (const float*)d_in[2];
  const float* Wproj = (const float*)d_in[3];
  float* out = (float*)d_out;

  unsigned char* ws = (unsigned char*)d_ws;
  const size_t SZ_X    = (size_t)Bn * Sn * En * 2;          // 8 MB
  const size_t SZ_WQKV = (size_t)3 * En * En * 2;           // 6 MB
  const size_t SZ_WPRJ = (size_t)En * En * 2;               // 2 MB
  const size_t SZ_HEAD = (size_t)Bn * Hn * Sn * Dn * 2;     // 8 MB

  unsigned short* xb     = (unsigned short*)(ws);
  unsigned short* wqkvt  = (unsigned short*)(ws + SZ_X);
  unsigned short* wprojt = (unsigned short*)(ws + SZ_X + SZ_WQKV);
  unsigned short* Qb     = (unsigned short*)(ws + SZ_X + SZ_WQKV + SZ_WPRJ);
  unsigned short* Kb     = (unsigned short*)((unsigned char*)Qb + SZ_HEAD);
  unsigned short* Vt     = (unsigned short*)((unsigned char*)Kb + SZ_HEAD);
  unsigned short* Ob     = (unsigned short*)((unsigned char*)Vt + SZ_HEAD);

  // 1. conversions
  {
    const int n = Bn * Sn * En;                     // 4,194,304
    k_f32_to_bf16<<<(n + 255) / 256, 256, 0, stream>>>(x, xb, n);
  }
  {
    const int n = 3 * En * En;                      // 3,145,728
    k_transpose_to_bf16<<<(n + 255) / 256, 256, 0, stream>>>(Wqkv, wqkvt, En, 3 * En);
  }
  {
    const int n = En * En;                          // 1,048,576
    k_transpose_to_bf16<<<(n + 255) / 256, 256, 0, stream>>>(Wproj, wprojt, En, En);
  }
  // 2. QKV GEMM: (4096/64)*(3072/64)=3072 waves, 8 waves/block
  k_gemm_qkv<<<384, 256, 0, stream>>>(xb, wqkvt, Qb, Kb, Vt);
  // 3. attention: B*H*(S/32) = 2048 single-wave blocks
  k_attention<<<Bn * Hn * (Sn / 32), 32, 0, stream>>>(Qb, Kb, Vt, Ob);
  // 4. projection: (4096/64)*(1024/64)=1024 waves, 8 waves/block
  k_gemm_proj<<<128, 256, 0, stream>>>(Ob, wprojt, out);
}
